// MyInformer_32315333935748
// MI455X (gfx1250) — compile-verified, matrix operand order checked by reading, not run on hardware
//
#include <hip/hip_runtime.h>
#include <hip/hip_bf16.h>

// ---------------------------------------------------------------------------
// Informer encoder forward for MI455X (gfx1250, wave32).
// GEMMs (~0.48 TFLOP hot path) run on v_wmma_f32_16x16x32_bf16 with tiles
// streamed into LDS by the Tensor Data Mover (tensor_load_to_lds, TENSORcnt),
// double-buffered so DMA overlaps the WMMA pipeline.
// ---------------------------------------------------------------------------

#define BATCH   128
#define SEQL    400
#define DMODEL  512
#define NHEADS  8
#define HEADD   64
#define NHID    512
#define NFEAT   10
#define UTOP    30      // FACTOR * ceil(ln(400)) = 5*6
#define USAMP   30

typedef __attribute__((ext_vector_type(16))) __bf16    v16bf;
typedef __attribute__((ext_vector_type(8)))  __bf16    v8bf;
typedef __attribute__((ext_vector_type(8)))  float     v8f;
typedef __attribute__((ext_vector_type(4)))  unsigned  u32x4;
typedef __attribute__((ext_vector_type(8)))  unsigned  u32x8;

__device__ inline __bf16 f2bf(float f) {
  union { float f; unsigned u; } in; in.f = f;
  unsigned u = in.u;
  unsigned r = u + 0x7FFFu + ((u >> 16) & 1u);   // round-to-nearest-even
  union { unsigned short s; __bf16 b; } out;
  out.s = (unsigned short)(r >> 16);
  return out.b;
}

__device__ inline unsigned ihash(unsigned x) {
  x ^= x >> 16; x *= 0x7feb352du;
  x ^= x >> 15; x *= 0x846ca68bu;
  x ^= x >> 16; return x;
}

// ---------------------------------------------------------------------------
// Tensor Data Mover descriptor helpers (CDNA5 ISA).
// 2D tile of 16-bit elements: tile_dim0=32 (one K step), tile_dim1=128 rows,
// LDS padding 4 dwords after every 16 dwords -> row stride 40 halfwords.
// ---------------------------------------------------------------------------
__device__ inline u32x8 tdm_group1(unsigned tensor_d0, unsigned tensor_d1,
                                   unsigned stride_d0) {
  u32x8 g;
  g[0] = (1u << 16)      // data_size = 1 -> 2 bytes
       | (1u << 20)      // pad_enable
       | (3u << 22)      // pad_interval: 16 dwords
       | (3u << 25);     // pad_amount:   4 dwords
  g[1] = (tensor_d0 & 0xFFFFu) << 16;                       // dim0 bits[15:0]
  g[2] = (tensor_d0 >> 16) | ((tensor_d1 & 0xFFFFu) << 16); // dim0 hi, dim1 lo
  g[3] = (tensor_d1 >> 16) | (32u << 16);                   // dim1 hi, tile_dim0=32
  g[4] = 128u;                                              // tile_dim1=128, tile_dim2=0
  g[5] = stride_d0;                                         // tensor_dim0_stride
  g[6] = 0u;
  g[7] = 0u;
  return g;
}

__device__ inline void tdm_issue(unsigned lds_addr, unsigned long long gaddr,
                                 u32x8 g1) {
  u32x4 g0;
  g0[0] = 1u;                                   // count=1, user descriptor
  g0[1] = lds_addr;                             // LDS byte address
  g0[2] = (unsigned)gaddr;                      // global_addr[31:0]
  g0[3] = (unsigned)(gaddr >> 32) | (2u << 30); // global_addr[56:32], type=2
  u32x4 zg = {0u, 0u, 0u, 0u};
  asm volatile("tensor_load_to_lds %0, %1, %2, %3"
               :: "s"(g0), "s"(g1), "s"(zg), "s"(zg)
               : "memory");
}

// ---------------------------------------------------------------------------
// f32 -> bf16 conversion (activations)
// ---------------------------------------------------------------------------
__global__ void cvt_bf16_kernel(const float* __restrict__ src,
                                __bf16* __restrict__ dst, int n) {
  int gid = blockIdx.x * blockDim.x + threadIdx.x;
  if (gid < n) dst[gid] = f2bf(src[gid]);
}

// f32 (K x N) -> bf16 transposed (N x K), once per launch per weight
__global__ void cvt_transpose_bf16_kernel(const float* __restrict__ src,
                                          __bf16* __restrict__ dst,
                                          int K, int N) {
  int gid = blockIdx.x * blockDim.x + threadIdx.x;
  if (gid >= K * N) return;
  int k = gid / N, n = gid % N;
  dst[(size_t)n * K + k] = f2bf(src[gid]);
}

// ---------------------------------------------------------------------------
// WMMA GEMM: C[M,N] = A[M,K](bf16) @ WT[N,K](bf16)^T + bias, optional GELU.
// Block tile 128x128, 8 wave32 waves (4 x 2), each wave a 32x64 patch via
// eight v_wmma_f32_16x16x32_bf16 per 32-deep K step. Tiles arrive via TDM
// (tensor_load_to_lds) into double-buffered, hardware-padded LDS.
// ---------------------------------------------------------------------------
template <int ACT>
__global__ __launch_bounds__(256)
void wmma_gemm_bf16(const __bf16* __restrict__ A, const __bf16* __restrict__ WT,
                    const float* __restrict__ bias, float* __restrict__ C,
                    int M, int N, int K) {
  constexpr int BM = 128, BN = 128, KT = 32;
  constexpr int ST = 40;                        // padded LDS row stride (bf16)
  __shared__ __align__(16) __bf16 sA[2][BM * ST];
  __shared__ __align__(16) __bf16 sB[2][BN * ST];

  const int tid  = threadIdx.x;
  const int lane = tid & 31;
  const int wid  = tid >> 5;
  const int wm   = wid & 3;                     // 4 waves along M: 32 rows each
  const int wn   = wid >> 2;                    // 2 waves along N: 64 cols each
  const int m0   = blockIdx.y * BM;
  const int n0   = blockIdx.x * BN;
  const int half = lane >> 4;
  const int l16  = lane & 15;

  const unsigned ldsA[2] = { (unsigned)(size_t)&sA[0][0], (unsigned)(size_t)&sA[1][0] };
  const unsigned ldsB[2] = { (unsigned)(size_t)&sB[0][0], (unsigned)(size_t)&sB[1][0] };
  const unsigned long long aBase =
      (unsigned long long)(size_t)A + (unsigned long long)m0 * K * 2ull;
  const unsigned long long bBase =
      (unsigned long long)(size_t)WT + (unsigned long long)n0 * K * 2ull;
  const u32x8 g1A = tdm_group1((unsigned)K, (unsigned)M, (unsigned)K);
  const u32x8 g1B = tdm_group1((unsigned)K, (unsigned)N, (unsigned)K);

  const int S = K / KT;
  if (wid == 0) {                               // wave 0 drives the TDM
    tdm_issue(ldsA[0], aBase, g1A);
    tdm_issue(ldsB[0], bBase, g1B);
  }

  v8f acc[2][4] = {};

  for (int s = 0; s < S; ++s) {
    if (wid == 0) {
      if (s + 1 < S) {                          // prefetch next tile pair
        unsigned long long go = (unsigned long long)(s + 1) * KT * 2ull;
        int nb = (s + 1) & 1;
        tdm_issue(ldsA[nb], aBase + go, g1A);
        tdm_issue(ldsB[nb], bBase + go, g1B);
        __builtin_amdgcn_s_wait_tensorcnt(2);   // tile s landed, s+1 in flight
      } else {
        __builtin_amdgcn_s_wait_tensorcnt(0);
      }
    }
    __syncthreads();

    const __bf16* cA = &sA[s & 1][0];
    const __bf16* cB = &sB[s & 1][0];

    // 16-bit A-fragment layout: lane = row, K split as half*8+{0..7} and
    // 16+half*8+{0..7}; B mirrored (lane = output column)
    v16bf af[2], bf[4];
    #pragma unroll
    for (int sm = 0; sm < 2; ++sm) {
      int ml = wm * 32 + sm * 16 + l16;
      v8bf lo = *(const v8bf*)&cA[ml * ST + half * 8];
      v8bf hi = *(const v8bf*)&cA[ml * ST + 16 + half * 8];
      af[sm] = __builtin_shufflevector(lo, hi, 0,1,2,3,4,5,6,7,8,9,10,11,12,13,14,15);
    }
    #pragma unroll
    for (int sn = 0; sn < 4; ++sn) {
      int nl = wn * 64 + sn * 16 + l16;
      v8bf lo = *(const v8bf*)&cB[nl * ST + half * 8];
      v8bf hi = *(const v8bf*)&cB[nl * ST + 16 + half * 8];
      bf[sn] = __builtin_shufflevector(lo, hi, 0,1,2,3,4,5,6,7,8,9,10,11,12,13,14,15);
    }
    #pragma unroll
    for (int sm = 0; sm < 2; ++sm)
      #pragma unroll
      for (int sn = 0; sn < 4; ++sn)
        acc[sm][sn] = __builtin_amdgcn_wmma_f32_16x16x32_bf16(
            false, af[sm], false, bf[sn], (short)0, acc[sm][sn], false, false);
    __syncthreads();
  }

  // epilogue: C 16x16 layout -> VGPR r holds M = half*8 + r, N = lane&15
  #pragma unroll
  for (int sm = 0; sm < 2; ++sm)
    #pragma unroll
    for (int sn = 0; sn < 4; ++sn)
      #pragma unroll
      for (int r = 0; r < 8; ++r) {
        int row = m0 + wm * 32 + sm * 16 + half * 8 + r;
        int col = n0 + wn * 64 + sn * 16 + l16;
        float v = acc[sm][sn][r] + (bias ? bias[col] : 0.0f);
        if (ACT == 1) v = 0.5f * v * (1.0f + erff(v * 0.70710678118f));
        C[(size_t)row * N + col] = v;
      }
}

// ---------------------------------------------------------------------------
// circular conv (K=3, wrap pad) over N_FEAT channels + sinusoidal pos embed
// ---------------------------------------------------------------------------
__global__ void conv_pos_kernel(const float* __restrict__ src,
                                const float* __restrict__ tw,
                                float* __restrict__ x) {
  int gid = blockIdx.x * blockDim.x + threadIdx.x;
  if (gid >= BATCH * SEQL * DMODEL) return;
  int c = gid % DMODEL;
  int l = (gid / DMODEL) % SEQL;
  int b = gid / (DMODEL * SEQL);
  float acc = 0.0f;
  #pragma unroll
  for (int t = 0; t < 3; ++t) {
    int lp = l + t - 1;
    lp = (lp < 0) ? lp + SEQL : (lp >= SEQL ? lp - SEQL : lp);
    const float* s = src + ((size_t)b * SEQL + lp) * NFEAT;
    #pragma unroll
    for (int i = 0; i < NFEAT; ++i)
      acc += s[i] * tw[(c * NFEAT + i) * 3 + t];
  }
  int j = c >> 1;
  float div = __expf((float)(2 * j) * (-9.210340371976184f / (float)DMODEL));
  float ang = (float)l * div;
  acc += (c & 1) ? __cosf(ang) : __sinf(ang);
  x[gid] = acc;
}

// ---------------------------------------------------------------------------
// ProbSparse: sparsity measure M = max(QK_s) - mean(QK_s) on sampled keys
// q,k layout: (B, L, H, D) flat
// ---------------------------------------------------------------------------
__global__ void prob_measure_kernel(const float* __restrict__ q,
                                    const float* __restrict__ k,
                                    float* __restrict__ M, int seed) {
  int gid = blockIdx.x * blockDim.x + threadIdx.x;
  if (gid >= BATCH * NHEADS * SEQL) return;
  int l = gid % SEQL;
  int h = (gid / SEQL) % NHEADS;
  int b = gid / (SEQL * NHEADS);
  const float* qp = q + (((size_t)b * SEQL + l) * NHEADS + h) * HEADD;
  float mx = -3.4e38f, sm = 0.0f;
  for (int s = 0; s < USAMP; ++s) {
    unsigned idx = ihash((unsigned)(l * 1315423911u) ^ (unsigned)(s * 2654435761u)
                         ^ (unsigned)(seed * 97531u)) % SEQL;
    const float* kp = k + (((size_t)b * SEQL + idx) * NHEADS + h) * HEADD;
    float d = 0.0f;
    #pragma unroll 8
    for (int t = 0; t < HEADD; ++t) d += qp[t] * kp[t];
    mx = fmaxf(mx, d);
    sm += d;
  }
  M[gid] = mx - sm * (1.0f / (float)SEQL);
}

// top-UTOP selection per (b,h): iterative parallel argmax
__global__ __launch_bounds__(128)
void topk_kernel(const float* __restrict__ M, int* __restrict__ Mtop) {
  __shared__ float sm[SEQL];
  __shared__ float rv[128];
  __shared__ int   ri[128];
  int bh = blockIdx.x;
  int tid = threadIdx.x;
  for (int i = tid; i < SEQL; i += 128) sm[i] = M[(size_t)bh * SEQL + i];
  __syncthreads();
  for (int it = 0; it < UTOP; ++it) {
    float bv = -3.4e38f; int bi = 0;
    for (int i = tid; i < SEQL; i += 128)
      if (sm[i] > bv) { bv = sm[i]; bi = i; }
    rv[tid] = bv; ri[tid] = bi;
    __syncthreads();
    if (tid == 0) {
      float gv = rv[0]; int gi = ri[0];
      for (int t = 1; t < 128; ++t)
        if (rv[t] > gv) { gv = rv[t]; gi = ri[t]; }
      Mtop[bh * UTOP + it] = gi;
      sm[gi] = -3.4e38f;
    }
    __syncthreads();
  }
}

// reduced attention: scores, causal mask, softmax, attn @ V
__global__ __launch_bounds__(128)
void sparse_attn_kernel(const float* __restrict__ q, const float* __restrict__ k,
                        const float* __restrict__ v, const int* __restrict__ Mtop,
                        float* __restrict__ ctxTop) {
  __shared__ float sc[SEQL];
  __shared__ float red[128];
  int blk = blockIdx.x;                 // (b*H + h)*UTOP + ui
  int ui = blk % UTOP;
  int bh = blk / UTOP;
  int h = bh % NHEADS, b = bh / NHEADS;
  int tid = threadIdx.x;
  int qidx = Mtop[bh * UTOP + ui];
  const float* qp = q + (((size_t)b * SEQL + qidx) * NHEADS + h) * HEADD;
  const float scale = 0.125f;           // 1/sqrt(64)
  for (int kk = tid; kk < SEQL; kk += 128) {
    const float* kp = k + (((size_t)b * SEQL + kk) * NHEADS + h) * HEADD;
    float d = 0.0f;
    #pragma unroll 8
    for (int t = 0; t < HEADD; ++t) d += qp[t] * kp[t];
    sc[kk] = (kk > qidx) ? -3.4e38f : d * scale;
  }
  __syncthreads();
  float lm = -3.4e38f;
  for (int kk = tid; kk < SEQL; kk += 128) lm = fmaxf(lm, sc[kk]);
  red[tid] = lm; __syncthreads();
  for (int s = 64; s > 0; s >>= 1) {
    if (tid < s) red[tid] = fmaxf(red[tid], red[tid + s]);
    __syncthreads();
  }
  float mx = red[0]; __syncthreads();
  float ls = 0.0f;
  for (int kk = tid; kk < SEQL; kk += 128) {
    float e = __expf(sc[kk] - mx);
    sc[kk] = e; ls += e;
  }
  __syncthreads();
  red[tid] = ls; __syncthreads();
  for (int s = 64; s > 0; s >>= 1) {
    if (tid < s) red[tid] += red[tid + s];
    __syncthreads();
  }
  float inv = 1.0f / red[0]; __syncthreads();
  if (tid < HEADD) {
    float acc = 0.0f;
    for (int kk = 0; kk < SEQL; ++kk)
      acc += sc[kk] * v[(((size_t)b * SEQL + kk) * NHEADS + h) * HEADD + tid];
    ctxTop[(size_t)blk * HEADD + tid] = acc * inv;
  }
}

// context = cumsum(v) along sequence; ctx stored as (B, L, H, D)
__global__ void cumsum_kernel(const float* __restrict__ v, float* __restrict__ ctx) {
  int gid = blockIdx.x * blockDim.x + threadIdx.x;
  if (gid >= BATCH * NHEADS * HEADD) return;
  int d = gid % HEADD;
  int h = (gid / HEADD) % NHEADS;
  int b = gid / (HEADD * NHEADS);
  float acc = 0.0f;
  for (int l = 0; l < SEQL; ++l) {
    size_t off = (((size_t)b * SEQL + l) * NHEADS + h) * HEADD + d;
    acc += v[off];
    ctx[off] = acc;
  }
}

__global__ void scatter_kernel(const float* __restrict__ ctxTop,
                               const int* __restrict__ Mtop,
                               float* __restrict__ ctx) {
  int gid = blockIdx.x * blockDim.x + threadIdx.x;
  if (gid >= BATCH * NHEADS * UTOP * HEADD) return;
  int d  = gid % HEADD;
  int ui = (gid / HEADD) % UTOP;
  int bh = gid / (HEADD * UTOP);
  int h = bh % NHEADS, b = bh / NHEADS;
  int qidx = Mtop[bh * UTOP + ui];
  ctx[(((size_t)b * SEQL + qidx) * NHEADS + h) * HEADD + d] = ctxTop[(size_t)gid];
}

// ---------------------------------------------------------------------------
// out = LayerNorm(x + y) * g + b   (y may be null => plain LayerNorm(x))
// ---------------------------------------------------------------------------
__global__ __launch_bounds__(256)
void add_ln_kernel(const float* __restrict__ x, const float* __restrict__ y,
                   const float* __restrict__ g, const float* __restrict__ b,
                   float* __restrict__ out) {
  __shared__ float s1[256], s2[256];
  int row = blockIdx.x, tid = threadIdx.x;
  const float* xr = x + (size_t)row * DMODEL;
  float v0 = xr[tid], v1 = xr[tid + 256];
  if (y) { v0 += y[(size_t)row * DMODEL + tid]; v1 += y[(size_t)row * DMODEL + tid + 256]; }
  s1[tid] = v0 + v1;
  s2[tid] = v0 * v0 + v1 * v1;
  __syncthreads();
  for (int s = 128; s > 0; s >>= 1) {
    if (tid < s) { s1[tid] += s1[tid + s]; s2[tid] += s2[tid + s]; }
    __syncthreads();
  }
  float mu  = s1[0] * (1.0f / DMODEL);
  float var = s2[0] * (1.0f / DMODEL) - mu * mu;
  float rs  = rsqrtf(var + 1e-5f);
  out[(size_t)row * DMODEL + tid]       = (v0 - mu) * rs * g[tid] + b[tid];
  out[(size_t)row * DMODEL + tid + 256] = (v1 - mu) * rs * g[tid + 256] + b[tid + 256];
}

// output head: sigmoid(x @ out_w + out_b)
__global__ void out_proj_kernel(const float* __restrict__ x,
                                const float* __restrict__ w,
                                const float* __restrict__ b,
                                float* __restrict__ out) {
  int gid = blockIdx.x * blockDim.x + threadIdx.x;
  if (gid >= BATCH * SEQL * NFEAT) return;
  int o = gid % NFEAT;
  int row = gid / NFEAT;
  const float* xr = x + (size_t)row * DMODEL;
  float acc = b[o];
  for (int t = 0; t < DMODEL; ++t) acc += xr[t] * w[t * NFEAT + o];
  out[gid] = 1.0f / (1.0f + __expf(-acc));
}

// ---------------------------------------------------------------------------
// host side
// ---------------------------------------------------------------------------
static void* ws_take(char*& p, size_t bytes) {
  void* r = (void*)p;
  p += (bytes + 255) & ~(size_t)255;
  return r;
}

static void launch_gemm(const __bf16* A, const __bf16* WT, const float* bias,
                        float* C, int M, int N, int K, int act, hipStream_t s) {
  dim3 g(N / 128, M / 128), blk(256);
  if (act == 1) wmma_gemm_bf16<1><<<g, blk, 0, s>>>(A, WT, bias, C, M, N, K);
  else          wmma_gemm_bf16<0><<<g, blk, 0, s>>>(A, WT, bias, C, M, N, K);
}

static void launch_cvt(const float* src, __bf16* dst, int n, hipStream_t s) {
  cvt_bf16_kernel<<<(n + 255) / 256, 256, 0, s>>>(src, dst, n);
}

static void launch_cvt_tr(const float* src, __bf16* dst, int K, int N, hipStream_t s) {
  int n = K * N;
  cvt_transpose_bf16_kernel<<<(n + 255) / 256, 256, 0, s>>>(src, dst, K, N);
}

extern "C" void kernel_launch(void* const* d_in, const int* in_sizes, int n_in,
                              void* d_out, int out_size, void* d_ws, size_t ws_size,
                              hipStream_t stream) {
  (void)in_sizes; (void)n_in; (void)out_size; (void)ws_size;
  const float* src = (const float*)d_in[0];
  // d_in[1] = attn_mask (unused scalar)
  // params flattened by pytree (sorted dict keys):
  //  per layer: bk bo bq bv c1_b c1_w c2_b c2_w n1_b n1_g n2_b n2_g wk wo wq wv
  struct Layer {
    const float *wq,*bq,*wk,*bk,*wv,*bv,*wo,*bo;
    const float *c1w,*c1b,*c2w,*c2b,*n1g,*n1b,*n2g,*n2b;
  } L[3];
  for (int i = 0; i < 3; ++i) {
    int base = 2 + 16 * i;
    L[i].bk  = (const float*)d_in[base + 0];
    L[i].bo  = (const float*)d_in[base + 1];
    L[i].bq  = (const float*)d_in[base + 2];
    L[i].bv  = (const float*)d_in[base + 3];
    L[i].c1b = (const float*)d_in[base + 4];
    L[i].c1w = (const float*)d_in[base + 5];
    L[i].c2b = (const float*)d_in[base + 6];
    L[i].c2w = (const float*)d_in[base + 7];
    L[i].n1b = (const float*)d_in[base + 8];
    L[i].n1g = (const float*)d_in[base + 9];
    L[i].n2b = (const float*)d_in[base + 10];
    L[i].n2g = (const float*)d_in[base + 11];
    L[i].wk  = (const float*)d_in[base + 12];
    L[i].wo  = (const float*)d_in[base + 13];
    L[i].wq  = (const float*)d_in[base + 14];
    L[i].wv  = (const float*)d_in[base + 15];
  }
  const float* norm_b  = (const float*)d_in[50];
  const float* norm_g  = (const float*)d_in[51];
  const float* out_b   = (const float*)d_in[52];
  const float* out_w   = (const float*)d_in[53];
  const float* token_w = (const float*)d_in[54];

  const int M  = BATCH * SEQL;            // 51200 rows
  const size_t actF = (size_t)M * DMODEL; // f32 activation elems
  char* wp = (char*)d_ws;
  float*  X    = (float*)ws_take(wp, actF * 4);
  float*  Q    = (float*)ws_take(wp, actF * 4);
  float*  Kb   = (float*)ws_take(wp, actF * 4);
  float*  Vb   = (float*)ws_take(wp, actF * 4);
  float*  CTX  = (float*)ws_take(wp, actF * 4);
  __bf16* ABF  = (__bf16*)ws_take(wp, actF * 2);
  float*  Mb   = (float*)ws_take(wp, (size_t)BATCH * NHEADS * SEQL * 4);
  int*    Mtop = (int*)ws_take(wp, (size_t)BATCH * NHEADS * UTOP * 4);
  float*  CTop = (float*)ws_take(wp, (size_t)BATCH * NHEADS * UTOP * HEADD * 4);
  __bf16* WBF[3][6];
  for (int i = 0; i < 3; ++i)
    for (int j = 0; j < 6; ++j)
      WBF[i][j] = (__bf16*)ws_take(wp, (size_t)DMODEL * DMODEL * 2);

  // transpose-convert all weights to bf16 (N x K) once per launch
  for (int i = 0; i < 3; ++i) {
    launch_cvt_tr(L[i].wq,  WBF[i][0], DMODEL, DMODEL, stream);
    launch_cvt_tr(L[i].wk,  WBF[i][1], DMODEL, DMODEL, stream);
    launch_cvt_tr(L[i].wv,  WBF[i][2], DMODEL, DMODEL, stream);
    launch_cvt_tr(L[i].wo,  WBF[i][3], DMODEL, DMODEL, stream);
    launch_cvt_tr(L[i].c1w, WBF[i][4], DMODEL, NHID,   stream);
    launch_cvt_tr(L[i].c2w, WBF[i][5], NHID,   DMODEL, stream);
  }

  // token embedding (circular conv) + positional encoding
  {
    int n = M * DMODEL;
    conv_pos_kernel<<<(n + 255) / 256, 256, 0, stream>>>(src, token_w, X);
  }

  for (int li = 0; li < 3; ++li) {
    // Q, K, V projections (WMMA + TDM)
    launch_cvt(X, ABF, (int)actF, stream);
    launch_gemm(ABF, WBF[li][0], L[li].bq, Q,  M, DMODEL, DMODEL, 0, stream);
    launch_gemm(ABF, WBF[li][1], L[li].bk, Kb, M, DMODEL, DMODEL, 0, stream);
    launch_gemm(ABF, WBF[li][2], L[li].bv, Vb, M, DMODEL, DMODEL, 0, stream);

    // ProbSparse attention
    {
      int n = BATCH * NHEADS * SEQL;
      prob_measure_kernel<<<(n + 255) / 256, 256, 0, stream>>>(Q, Kb, Mb, li + 1);
    }
    topk_kernel<<<BATCH * NHEADS, 128, 0, stream>>>(Mb, Mtop);
    sparse_attn_kernel<<<BATCH * NHEADS * UTOP, 128, 0, stream>>>(Q, Kb, Vb, Mtop, CTop);
    {
      int n = BATCH * NHEADS * HEADD;
      cumsum_kernel<<<(n + 255) / 256, 256, 0, stream>>>(Vb, CTX);
    }
    {
      int n = BATCH * NHEADS * UTOP * HEADD;
      scatter_kernel<<<(n + 255) / 256, 256, 0, stream>>>(CTop, Mtop, CTX);
    }

    // output projection + residual LN
    launch_cvt(CTX, ABF, (int)actF, stream);
    launch_gemm(ABF, WBF[li][3], L[li].bo, Q, M, DMODEL, DMODEL, 0, stream);
    add_ln_kernel<<<M, 256, 0, stream>>>(X, Q, L[li].n1g, L[li].n1b, X);

    // FFN: gelu(x@c1+b1)@c2+b2, residual LN
    launch_cvt(X, ABF, (int)actF, stream);
    launch_gemm(ABF, WBF[li][4], L[li].c1b, Kb, M, NHID, DMODEL, 1, stream);
    launch_cvt(Kb, ABF, (int)actF, stream);
    launch_gemm(ABF, WBF[li][5], L[li].c2b, Q, M, DMODEL, NHID, 0, stream);
    add_ln_kernel<<<M, 256, 0, stream>>>(X, Q, L[li].n2g, L[li].n2b, X);
  }

  // final norm + sigmoid head
  add_ln_kernel<<<M, 256, 0, stream>>>(X, nullptr, norm_g, norm_b, X);
  {
    int n = M * NFEAT;
    out_proj_kernel<<<(n + 255) / 256, 256, 0, stream>>>(X, out_w, out_b, (float*)d_out);
  }
}
